// SAttention_76742475645472
// MI455X (gfx1250) — matrix operand order repaired and checked by reading, not compiled, not run on hardware
//
#include <hip/hip_runtime.h>
#include <hip/hip_bf16.h>
#include <cstdint>

// ---------------------------------------------------------------------------
// Spiking attention (SAttention), MI455X / gfx1250, wave32 + WMMA f16 +
// async global->LDS copies (GLOBAL_LOAD_ASYNC_TO_LDS_B128, ASYNCcnt).
// T=8 B=8 N=512 C=768 H=12 D=64. Spikes/IF-accumulators are exact in f16/i8.
// GEMMs: v_wmma_f32_16x16x32_f16, 128-wide LDS double-buffered tiles.
// ---------------------------------------------------------------------------

typedef __attribute__((ext_vector_type(16))) _Float16 v16h;
typedef __attribute__((ext_vector_type(8)))  _Float16 v8h;
typedef __attribute__((ext_vector_type(8)))  float    v8f;

#define T_STEPS 8
#define NN      512
#define CC      768
#define HH      12
#define DD      64
#define BN      4096      // B*N
#define C3      2304      // 3*C
#define BH      96        // B*H
#define LDK     40        // LDS row stride (f16 elems): 32 data + 8 pad = 80B
#define ABUF    (128 * LDK)

// ---------------- CDNA5 async global->LDS copy (16B per lane) --------------
__device__ __forceinline__ void async_b128(void* lds, const void* g) {
  asm volatile("global_load_async_to_lds_b128 %0, %1, off"
               :: "v"((uint32_t)(uintptr_t)lds), "v"((uint64_t)(uintptr_t)g)
               : "memory");
}
__device__ __forceinline__ void wait_async0() {
  asm volatile("s_wait_asynccnt 0x0" ::: "memory");
}

// ---------------- IF neuron step (q_threshold=1, borrow=0) -----------------
// acc is an exact small integer in [-8,7]; jnp.round is a no-op on it.
__device__ __forceinline__ float if_step_i(float& q, int& acc, float x,
                                           int posMax, int negMin) {
  q += x;
  float out = 0.0f;
  if ((q - 1.0f >= 0.0f) && (acc < posMax))      out = 1.0f;
  else if ((q < 0.0f) && (acc > negMin))         out = -1.0f;
  q   -= out;
  acc += (int)out;
  return out;
}

// ---------------- WMMA helpers (CDNA5 layouts, wave32) ---------------------
__device__ __forceinline__ v8f wmma16(v16h a, v16h b, v8f c) {
  return __builtin_amdgcn_wmma_f32_16x16x32_f16(false, a, false, b,
                                                (short)0, c, false, false);
}
// A fragment 16x32 f16: lane l row m=l&15, k = (e/8)*16 + (l>>4)*8 + e%8.
__device__ __forceinline__ v16h frag_a(const _Float16* tile, int mbase, int lane) {
  const int m = mbase + (lane & 15);
  const int hf = lane >> 4;
  const _Float16* r = tile + m * LDK;
  v8h lo = *(const v8h*)(r + hf * 8);
  v8h hi = *(const v8h*)(r + 16 + hf * 8);
  v16h f;
#pragma unroll
  for (int i = 0; i < 8; ++i) { f[i] = lo[i]; f[i + 8] = hi[i]; }
  return f;
}
// B fragment 32x16 f16: lane l col n=l&15, k = (e/8)*16 + (l>>4)*8 + e%8.
__device__ __forceinline__ v16h frag_b(const _Float16* tile, int nbase, int lane) {
  const int n = nbase + (lane & 15);
  const int hf = lane >> 4;
  const _Float16* r = tile + n * LDK;
  v8h lo = *(const v8h*)(r + hf * 8);
  v8h hi = *(const v8h*)(r + 16 + hf * 8);
  v16h f;
#pragma unroll
  for (int i = 0; i < 8; ++i) { f[i] = lo[i]; f[i + 8] = hi[i]; }
  return f;
}

// ---------------- utility kernels ------------------------------------------
__global__ void k_fill_f32(float* p, float v, size_t n) {
  size_t i = (size_t)blockIdx.x * blockDim.x + threadIdx.x;
  if (i < n) p[i] = v;
}
__global__ void k_fill_i8(int8_t* p, int v, size_t n) {
  size_t i = (size_t)blockIdx.x * blockDim.x + threadIdx.x;
  if (i < n) p[i] = (int8_t)v;
}
__global__ void k_fill_f16(_Float16* p, float v, size_t n) {
  size_t i = (size_t)blockIdx.x * blockDim.x + threadIdx.x;
  if (i < n) p[i] = (_Float16)v;
}
__global__ void k_f32_to_f16(const float* __restrict__ s, _Float16* __restrict__ d, size_t n) {
  size_t i = (size_t)blockIdx.x * blockDim.x + threadIdx.x;
  if (i < n) d[i] = (_Float16)s[i];
}

// ---------------- K1: qkv = x_t @ w_qkv^T  (4096 x 2304 x 768) -------------
// Fused epilogue: q/k/v IF; emits f16 GEMM operands:
//  qs=q*s, qa=acc_q*s (row-major [BN,C]); k, kc=acc_k-k (row-major);
//  vT, vcT = v / (acc_v - v) TRANSPOSED to [B,H,D,N] so K4 stages contiguously.
__global__ __launch_bounds__(256) void k_qkv_gemm(
    const _Float16* __restrict__ x16, const _Float16* __restrict__ w16,
    float* __restrict__ stQ, int8_t* __restrict__ stA,
    _Float16* __restrict__ qs16, _Float16* __restrict__ qa16,
    _Float16* __restrict__ k16,  _Float16* __restrict__ kc16,
    _Float16* __restrict__ vT16, _Float16* __restrict__ vcT16) {
  __shared__ __align__(16) _Float16 As[2 * ABUF];
  __shared__ __align__(16) _Float16 Bs[2 * ABUF];
  const int tile_m = blockIdx.y, tile_n = blockIdx.x;
  const int tid = threadIdx.x, wid = tid >> 5, lane = tid & 31;
  const int wm = wid & 3, wn = wid >> 2;           // 4x2 waves of 32x64
  const int row0 = tile_m * 128, col0 = tile_n * 128;
  const int rA = tid >> 1, sA = (tid & 1) * 16;
  v8f acc[2][4];
#pragma unroll
  for (int i = 0; i < 2; ++i)
#pragma unroll
    for (int j = 0; j < 4; ++j) acc[i][j] = (v8f)0.0f;

  auto issue = [&](int kc, int bufI) {
    _Float16* Ad = As + bufI * ABUF + rA * LDK + sA;
    const _Float16* sa = x16 + (size_t)(row0 + rA) * CC + kc + sA;
    async_b128(Ad, sa); async_b128(Ad + 8, sa + 8);
    _Float16* Bd = Bs + bufI * ABUF + rA * LDK + sA;
    const _Float16* sb = w16 + (size_t)(col0 + rA) * CC + kc + sA;
    async_b128(Bd, sb); async_b128(Bd + 8, sb + 8);
  };
  const int NC4 = CC / 32;
  issue(0, 0);
  for (int ch = 0; ch < NC4; ++ch) {
    wait_async0();                 // chunk ch landed (only it is outstanding)
    __syncthreads();               // all waves landed; prev compute done
    if (ch + 1 < NC4) issue((ch + 1) * 32, (ch + 1) & 1);
    const _Float16* Ab = As + (ch & 1) * ABUF;
    const _Float16* Bb = Bs + (ch & 1) * ABUF;
    v16h af[2], bf[4];
#pragma unroll
    for (int mi = 0; mi < 2; ++mi) af[mi] = frag_a(Ab, wm * 32 + mi * 16, lane);
#pragma unroll
    for (int ni = 0; ni < 4; ++ni) bf[ni] = frag_b(Bb, wn * 64 + ni * 16, lane);
#pragma unroll
    for (int mi = 0; mi < 2; ++mi)
#pragma unroll
      for (int ni = 0; ni < 4; ++ni) acc[mi][ni] = wmma16(af[mi], bf[ni], acc[mi][ni]);
  }

  const int hf = lane >> 4, ncl = lane & 15;
#pragma unroll
  for (int mi = 0; mi < 2; ++mi)
#pragma unroll
    for (int ni = 0; ni < 4; ++ni) {
      const int col = col0 + wn * 64 + ni * 16 + ncl;  // [0,2304)
      const int s = col / CC;                          // uniform per 16-col tile
      const int hd = col % CC;
#pragma unroll
      for (int j = 0; j < 8; ++j) {
        const int row = row0 + wm * 32 + mi * 16 + hf * 8 + j;
        const size_t sidx = (size_t)row * C3 + col;
        float qst = stQ[sidx];
        int   ac  = stA[sidx];
        const float out = if_step_i(qst, ac, acc[mi][ni][j], 7, -8);
        stQ[sidx] = qst; stA[sidx] = (int8_t)ac;
        if (s == 0) {
          const size_t didx = (size_t)row * CC + hd;
          qs16[didx] = (_Float16)(out * 0.125f);
          qa16[didx] = (_Float16)((float)ac * 0.125f);
        } else if (s == 1) {
          const size_t didx = (size_t)row * CC + hd;
          k16[didx]  = (_Float16)out;
          kc16[didx] = (_Float16)(float)(ac - (int)out);
        } else {                                       // v: transposed [B,H,D,N]
          const int b2 = row >> 9, n2 = row & (NN - 1);
          const int h2 = hd >> 6, d2 = hd & (DD - 1);
          const size_t tix = (((size_t)(b2 * HH + h2) * DD + d2) * NN) + n2;
          vT16[tix]  = (_Float16)out;
          vcT16[tix] = (_Float16)(float)(ac - (int)out);
        }
      }
    }
}

// ---------------- K2: scores per head: [qa|qs] @ [k|kc]^T, K=128 -----------
__global__ __launch_bounds__(256) void k_score_gemm(
    const _Float16* __restrict__ qa16, const _Float16* __restrict__ qs16,
    const _Float16* __restrict__ k16,  const _Float16* __restrict__ kc16,
    float* __restrict__ aifQ, int8_t* __restrict__ aifA, float* __restrict__ smX) {
  __shared__ __align__(16) _Float16 As[2 * ABUF];
  __shared__ __align__(16) _Float16 Bs[2 * ABUF];
  const int bh = blockIdx.z, b = bh / HH, h = bh % HH;
  const int tile_m = blockIdx.y, tile_n = blockIdx.x;
  const int tid = threadIdx.x, wid = tid >> 5, lane = tid & 31;
  const int wm = wid & 3, wn = wid >> 2;
  const int row0 = tile_m * 128, col0 = tile_n * 128;
  const size_t hbase = (size_t)b * NN * CC + (size_t)h * DD;
  const int rA = tid >> 1, sA = (tid & 1) * 16;
  v8f acc[2][4];
#pragma unroll
  for (int i = 0; i < 2; ++i)
#pragma unroll
    for (int j = 0; j < 4; ++j) acc[i][j] = (v8f)0.0f;

  auto issue = [&](int ch, int bufI) {
    const _Float16* Aseg = (ch < 2) ? qa16 : qs16;
    const _Float16* Bseg = (ch < 2) ? k16 : kc16;
    const int koff = (ch & 1) * 32;
    _Float16* Ad = As + bufI * ABUF + rA * LDK + sA;
    const _Float16* sa = Aseg + hbase + (size_t)(row0 + rA) * CC + koff + sA;
    async_b128(Ad, sa); async_b128(Ad + 8, sa + 8);
    _Float16* Bd = Bs + bufI * ABUF + rA * LDK + sA;
    const _Float16* sb = Bseg + hbase + (size_t)(col0 + rA) * CC + koff + sA;
    async_b128(Bd, sb); async_b128(Bd + 8, sb + 8);
  };
  issue(0, 0);
  for (int ch = 0; ch < 4; ++ch) {
    wait_async0();
    __syncthreads();
    if (ch + 1 < 4) issue(ch + 1, (ch + 1) & 1);
    const _Float16* Ab = As + (ch & 1) * ABUF;
    const _Float16* Bb = Bs + (ch & 1) * ABUF;
    v16h af[2], bf[4];
#pragma unroll
    for (int mi = 0; mi < 2; ++mi) af[mi] = frag_a(Ab, wm * 32 + mi * 16, lane);
#pragma unroll
    for (int ni = 0; ni < 4; ++ni) bf[ni] = frag_b(Bb, wn * 64 + ni * 16, lane);
#pragma unroll
    for (int mi = 0; mi < 2; ++mi)
#pragma unroll
      for (int ni = 0; ni < 4; ++ni) acc[mi][ni] = wmma16(af[mi], bf[ni], acc[mi][ni]);
  }

  const int hf = lane >> 4, ncl = lane & 15;
#pragma unroll
  for (int mi = 0; mi < 2; ++mi)
#pragma unroll
    for (int ni = 0; ni < 4; ++ni) {
      const int col = col0 + wn * 64 + ni * 16 + ncl;
#pragma unroll
      for (int j = 0; j < 8; ++j) {
        const int row = row0 + wm * 32 + mi * 16 + hf * 8 + j;
        const size_t idx = ((size_t)bh * NN + row) * NN + col;
        float qst = aifQ[idx];
        int   ac  = aifA[idx];
        const float out = if_step_i(qst, ac, acc[mi][ni][j], 7, -8);
        aifQ[idx] = qst; aifA[idx] = (int8_t)ac;
        smX[idx] += out;
      }
    }
}

// ---------------- K3: spiking softmax + softmax-IF (one row / block) -------
// sif accumulator state IS sacc16 (f16-exact small ints), also the K4 operand.
__global__ __launch_bounds__(256) void k_softmax_if(
    const float* __restrict__ smX, float* __restrict__ smY,
    float* __restrict__ sifQ, _Float16* __restrict__ sacc16,
    _Float16* __restrict__ a16, float tscale) {
  __shared__ float red[8];
  const size_t rb = (size_t)blockIdx.x * NN;
  const int tid = threadIdx.x;
  const float x0 = smX[rb + tid], x1 = smX[rb + tid + 256];

  float m = fmaxf(x0, x1);
#pragma unroll
  for (int o = 16; o; o >>= 1) m = fmaxf(m, __shfl_xor(m, o, 32));
  if ((tid & 31) == 0) red[tid >> 5] = m;
  __syncthreads();
  m = red[0];
#pragma unroll
  for (int i = 1; i < 8; ++i) m = fmaxf(m, red[i]);
  __syncthreads();

  const float e0 = __expf(x0 - m), e1 = __expf(x1 - m);
  float s = e0 + e1;
#pragma unroll
  for (int o = 16; o; o >>= 1) s += __shfl_xor(s, o, 32);
  if ((tid & 31) == 0) red[tid >> 5] = s;
  __syncthreads();
  s = red[0];
#pragma unroll
  for (int i = 1; i < 8; ++i) s += red[i];
  const float sc = tscale / s;

#pragma unroll
  for (int half = 0; half < 2; ++half) {
    const size_t idx = rb + tid + half * 256;
    const float Y = (half ? e1 : e0) * sc;
    const float d = Y - smY[idx];
    smY[idx] = Y;
    float qst = sifQ[idx];
    int   ac  = (int)(float)sacc16[idx];
    const float out = if_step_i(qst, ac, d, 7, 0);     // sym=False: neg_min=0
    sifQ[idx] = qst;
    sacc16[idx] = (_Float16)(float)ac;
    a16[idx] = (_Float16)out;
  }
}

// ---------------- K4: out = [sacc|a] @ [vT ; vcT]^T, per head 512x64 -------
// B source now [B,H,D,N] (transposed at K1), so staging is contiguous async.
__global__ __launch_bounds__(256) void k_av_gemm(
    const _Float16* __restrict__ sacc16, const _Float16* __restrict__ a16,
    const _Float16* __restrict__ vT16,   const _Float16* __restrict__ vcT16,
    float* __restrict__ oifQ, int8_t* __restrict__ oifA, _Float16* __restrict__ o16) {
  __shared__ __align__(16) _Float16 As[2 * ABUF];
  __shared__ __align__(16) _Float16 Bs[2 * (64 * LDK)];
  const int bh = blockIdx.y, b = bh / HH, h = bh % HH;
  const int tile_m = blockIdx.x;
  const int tid = threadIdx.x, wid = tid >> 5, lane = tid & 31;
  const int wm = wid & 3, wn = wid >> 2;               // 4x2 waves of 32x32
  const int row0 = tile_m * 128;
  const size_t arb = (size_t)bh * NN * NN;
  const size_t vtb = (size_t)bh * DD * NN;
  const int rA = tid >> 1, sA = (tid & 1) * 16;
  const int dRow = tid >> 2, sB = (tid & 3) * 8;       // 64 rows x 4 thr x 8 f16
  v8f acc[2][2];
#pragma unroll
  for (int i = 0; i < 2; ++i)
#pragma unroll
    for (int j = 0; j < 2; ++j) acc[i][j] = (v8f)0.0f;

  auto issue = [&](int ch, int bufI) {
    const _Float16* Aseg = (ch < 16) ? sacc16 : a16;
    const _Float16* Bseg = (ch < 16) ? vT16 : vcT16;
    const int koff = (ch & 15) * 32;
    _Float16* Ad = As + bufI * ABUF + rA * LDK + sA;
    const _Float16* sa = Aseg + arb + (size_t)(row0 + rA) * NN + koff + sA;
    async_b128(Ad, sa); async_b128(Ad + 8, sa + 8);
    _Float16* Bd = Bs + bufI * (64 * LDK) + dRow * LDK + sB;
    const _Float16* sb = Bseg + vtb + (size_t)dRow * NN + koff + sB;
    async_b128(Bd, sb);
  };
  issue(0, 0);
  for (int ch = 0; ch < 32; ++ch) {                    // K = 2 segs * 512
    wait_async0();
    __syncthreads();
    if (ch + 1 < 32) issue(ch + 1, (ch + 1) & 1);
    const _Float16* Ab = As + (ch & 1) * ABUF;
    const _Float16* Bb = Bs + (ch & 1) * (64 * LDK);
    v16h af[2], bf[2];
#pragma unroll
    for (int mi = 0; mi < 2; ++mi) af[mi] = frag_a(Ab, wm * 32 + mi * 16, lane);
#pragma unroll
    for (int ni = 0; ni < 2; ++ni) bf[ni] = frag_b(Bb, wn * 32 + ni * 16, lane);
#pragma unroll
    for (int mi = 0; mi < 2; ++mi)
#pragma unroll
      for (int ni = 0; ni < 2; ++ni) acc[mi][ni] = wmma16(af[mi], bf[ni], acc[mi][ni]);
  }

  const int hf = lane >> 4, ncl = lane & 15;
#pragma unroll
  for (int mi = 0; mi < 2; ++mi)
#pragma unroll
    for (int ni = 0; ni < 2; ++ni) {
      const int col = wn * 32 + ni * 16 + ncl;         // d in [0,64)
#pragma unroll
      for (int j = 0; j < 8; ++j) {
        const int row = row0 + wm * 32 + mi * 16 + hf * 8 + j;      // n
        const size_t idx = (size_t)(b * NN + row) * CC + h * DD + col; // [B,N,C]
        float qst = oifQ[idx];
        int   ac  = oifA[idx];
        const float out = if_step_i(qst, ac, acc[mi][ni][j], 7, -8);
        oifQ[idx] = qst; oifA[idx] = (int8_t)ac;
        o16[idx] = (_Float16)out;
      }
    }
}

// ---------------- K5: proj = o @ w_proj^T (4096x768x768) + IF -> d_out -----
__global__ __launch_bounds__(256) void k_proj_gemm(
    const _Float16* __restrict__ o16, const _Float16* __restrict__ wp16,
    float* __restrict__ pifQ, int8_t* __restrict__ pifA, float* __restrict__ outp) {
  __shared__ __align__(16) _Float16 As[2 * ABUF];
  __shared__ __align__(16) _Float16 Bs[2 * ABUF];
  const int tile_m = blockIdx.y, tile_n = blockIdx.x;
  const int tid = threadIdx.x, wid = tid >> 5, lane = tid & 31;
  const int wm = wid & 3, wn = wid >> 2;
  const int row0 = tile_m * 128, col0 = tile_n * 128;
  const int rA = tid >> 1, sA = (tid & 1) * 16;
  v8f acc[2][4];
#pragma unroll
  for (int i = 0; i < 2; ++i)
#pragma unroll
    for (int j = 0; j < 4; ++j) acc[i][j] = (v8f)0.0f;

  auto issue = [&](int kc, int bufI) {
    _Float16* Ad = As + bufI * ABUF + rA * LDK + sA;
    const _Float16* sa = o16 + (size_t)(row0 + rA) * CC + kc + sA;
    async_b128(Ad, sa); async_b128(Ad + 8, sa + 8);
    _Float16* Bd = Bs + bufI * ABUF + rA * LDK + sA;
    const _Float16* sb = wp16 + (size_t)(col0 + rA) * CC + kc + sA;
    async_b128(Bd, sb); async_b128(Bd + 8, sb + 8);
  };
  const int NC4 = CC / 32;
  issue(0, 0);
  for (int ch = 0; ch < NC4; ++ch) {
    wait_async0();
    __syncthreads();
    if (ch + 1 < NC4) issue((ch + 1) * 32, (ch + 1) & 1);
    const _Float16* Ab = As + (ch & 1) * ABUF;
    const _Float16* Bb = Bs + (ch & 1) * ABUF;
    v16h af[2], bf[4];
#pragma unroll
    for (int mi = 0; mi < 2; ++mi) af[mi] = frag_a(Ab, wm * 32 + mi * 16, lane);
#pragma unroll
    for (int ni = 0; ni < 4; ++ni) bf[ni] = frag_b(Bb, wn * 64 + ni * 16, lane);
#pragma unroll
    for (int mi = 0; mi < 2; ++mi)
#pragma unroll
      for (int ni = 0; ni < 4; ++ni) acc[mi][ni] = wmma16(af[mi], bf[ni], acc[mi][ni]);
  }

  const int hf = lane >> 4, ncl = lane & 15;
#pragma unroll
  for (int mi = 0; mi < 2; ++mi)
#pragma unroll
    for (int ni = 0; ni < 4; ++ni) {
      const int col = col0 + wn * 64 + ni * 16 + ncl;
#pragma unroll
      for (int j = 0; j < 8; ++j) {
        const int row = row0 + wm * 32 + mi * 16 + hf * 8 + j;
        const size_t idx = (size_t)row * CC + col;
        float qst = pifQ[idx];
        int   ac  = pifA[idx];
        const float out = if_step_i(qst, ac, acc[mi][ni][j], 7, -8);
        pifQ[idx] = qst; pifA[idx] = (int8_t)ac;
        outp[idx] = out;                               // f32 spike output
      }
    }
}

// ---------------------------------------------------------------------------
extern "C" void kernel_launch(void* const* d_in, const int* in_sizes, int n_in,
                              void* d_out, int out_size, void* d_ws, size_t ws_size,
                              hipStream_t stream) {
  const float* x     = (const float*)d_in[0];   // [T,B,N,C]
  const float* wqkv  = (const float*)d_in[1];   // [3C,C]
  const float* wproj = (const float*)d_in[2];   // [C,C]
  float* out = (float*)d_out;                   // [T,B,N,C] f32

  char* base = (char*)d_ws;
  size_t off = 0;
  auto alloc = [&](size_t bytes) -> void* {
    void* p = base + off;
    off = (off + bytes + 255) & ~(size_t)255;
    return p;
  };
  const size_t nBNC  = (size_t)BN * CC;         // 3,145,728
  const size_t nBN3C = (size_t)BN * C3;         // 9,437,184
  const size_t nBHNN = (size_t)BH * NN * NN;    // 25,165,824
  const size_t nBHDN = (size_t)BH * DD * NN;    // 3,145,728

  _Float16* x16  = (_Float16*)alloc((size_t)T_STEPS * nBNC * 2);
  _Float16* wq16 = (_Float16*)alloc((size_t)C3 * CC * 2);
  _Float16* wp16 = (_Float16*)alloc((size_t)CC * CC * 2);
  float*   stQ   = (float*)alloc(nBN3C * 4);
  int8_t*  stA   = (int8_t*)alloc(nBN3C);
  _Float16* qs16 = (_Float16*)alloc(nBNC * 2);
  _Float16* qa16 = (_Float16*)alloc(nBNC * 2);
  _Float16* k16  = (_Float16*)alloc(nBNC * 2);
  _Float16* kc16 = (_Float16*)alloc(nBNC * 2);
  _Float16* vT16 = (_Float16*)alloc(nBHDN * 2);
  _Float16* vcT16= (_Float16*)alloc(nBHDN * 2);
  float*   aifQ  = (float*)alloc(nBHNN * 4);
  int8_t*  aifA  = (int8_t*)alloc(nBHNN);
  float*   smX   = (float*)alloc(nBHNN * 4);
  float*   smY   = (float*)alloc(nBHNN * 4);
  float*   sifQ  = (float*)alloc(nBHNN * 4);
  _Float16* sacc16 = (_Float16*)alloc(nBHNN * 2);
  _Float16* a16  = (_Float16*)alloc(nBHNN * 2);
  float*   oifQ  = (float*)alloc(nBNC * 4);
  int8_t*  oifA  = (int8_t*)alloc(nBNC);
  float*   pifQ  = (float*)alloc(nBNC * 4);
  int8_t*  pifA  = (int8_t*)alloc(nBNC);
  _Float16* o16  = (_Float16*)alloc(nBNC * 2);
  (void)ws_size; (void)in_sizes; (void)n_in; (void)out_size;

  const int TH = 256;
  auto blocks = [](size_t n) { return (unsigned)((n + 255) / 256); };

  // ---- converts + deterministic state init (every call) ----
  k_f32_to_f16<<<blocks((size_t)T_STEPS * nBNC), TH, 0, stream>>>(x, x16, (size_t)T_STEPS * nBNC);
  k_f32_to_f16<<<blocks((size_t)C3 * CC), TH, 0, stream>>>(wqkv, wq16, (size_t)C3 * CC);
  k_f32_to_f16<<<blocks((size_t)CC * CC), TH, 0, stream>>>(wproj, wp16, (size_t)CC * CC);

  k_fill_f32<<<blocks(nBN3C), TH, 0, stream>>>(stQ, 0.5f, nBN3C);
  k_fill_i8 <<<blocks(nBN3C), TH, 0, stream>>>(stA, 0, nBN3C);
  k_fill_f32<<<blocks(nBHNN), TH, 0, stream>>>(aifQ, 0.5f, nBHNN);
  k_fill_i8 <<<blocks(nBHNN), TH, 0, stream>>>(aifA, 0, nBHNN);
  k_fill_f32<<<blocks(nBHNN), TH, 0, stream>>>(smX, 0.0f, nBHNN);
  k_fill_f32<<<blocks(nBHNN), TH, 0, stream>>>(smY, 0.0f, nBHNN);
  k_fill_f32<<<blocks(nBHNN), TH, 0, stream>>>(sifQ, 0.5f, nBHNN);
  k_fill_f16<<<blocks(nBHNN), TH, 0, stream>>>(sacc16, 0.0f, nBHNN);
  k_fill_f32<<<blocks(nBNC), TH, 0, stream>>>(oifQ, 0.5f, nBNC);
  k_fill_i8 <<<blocks(nBNC), TH, 0, stream>>>(oifA, 0, nBNC);
  k_fill_f32<<<blocks(nBNC), TH, 0, stream>>>(pifQ, 0.5f, nBNC);
  k_fill_i8 <<<blocks(nBNC), TH, 0, stream>>>(pifA, 0, nBNC);

  // ---- timestep loop ----
  for (int tIdx = 0; tIdx < T_STEPS; ++tIdx) {
    const float tval = (float)(tIdx + 1);
    const float tscale = (tval / 6.0f < 1.0f) ? (tval / 6.0f) : 1.0f;

    k_qkv_gemm<<<dim3(C3 / 128, BN / 128), TH, 0, stream>>>(
        x16 + (size_t)tIdx * nBNC, wq16, stQ, stA,
        qs16, qa16, k16, kc16, vT16, vcT16);

    k_score_gemm<<<dim3(NN / 128, NN / 128, BH), TH, 0, stream>>>(
        qa16, qs16, k16, kc16, aifQ, aifA, smX);

    k_softmax_if<<<BH * NN, TH, 0, stream>>>(
        smX, smY, sifQ, sacc16, a16, tscale);

    k_av_gemm<<<dim3(NN / 128, BH), TH, 0, stream>>>(
        sacc16, a16, vT16, vcT16, oifQ, oifA, o16);

    k_proj_gemm<<<dim3(CC / 128, BN / 128), TH, 0, stream>>>(
        o16, wp16, pifQ, pifA, out + (size_t)tIdx * nBNC);
  }
}
